// NestedCell2_59493886984654
// MI455X (gfx1250) — matrix operand order repaired and checked by reading, not compiled, not run on hardware
//
#include <hip/hip_runtime.h>

typedef __attribute__((ext_vector_type(16))) _Float16 v16h;
typedef __attribute__((ext_vector_type(8)))  _Float16 v8h;
typedef __attribute__((ext_vector_type(8)))  float    v8f;

#define Bb   8
#define Nn   1024
#define Ff   64
#define Hh   4
#define Cc   32
#define HIN  128
#define HID  128
#define BN   (Bb*Nn)      /* 8192 */
#define NEGINF (-1.0e9f)

// ---------------- fragment loaders (documented CDNA5 wave32 layouts) -----------
// A 16x32 f16: lanes 0-15: M=lane, K in {0..7,16..23}; lanes 16-31: M=lane-16,
// K in {8..15,24..31}.  ld (in halfs) must be a multiple of 8 for aligned b128.
static __device__ __forceinline__ v16h frag_a_lds(const _Float16* A, int lane, int ld) {
  int m  = lane & 15;
  int k0 = (lane < 16) ? 0 : 8;
  v8h lo = *(const v8h*)(A + m*ld + k0);
  v8h hi = *(const v8h*)(A + m*ld + k0 + 16);
  v16h r;
#pragma unroll
  for (int t = 0; t < 8; ++t) { r[t] = lo[t]; r[8+t] = hi[t]; }
  return r;
}
// B 32x16 f16 staged transposed as Bt[n][k]: lanes 0-15: N=lane, K in {0..7,16..23};
// lanes 16-31: N=lane-16, K in {8..15,24..31}.
static __device__ __forceinline__ v16h frag_b_lds(const _Float16* Bt, int lane, int ld) {
  int n  = lane & 15;
  int k0 = (lane < 16) ? 0 : 8;
  v8h lo = *(const v8h*)(Bt + n*ld + k0);
  v8h hi = *(const v8h*)(Bt + n*ld + k0 + 16);
  v16h r;
#pragma unroll
  for (int t = 0; t < 8; ++t) { r[t] = lo[t]; r[8+t] = hi[t]; }
  return r;
}

// ------------- generic 64x64-tile WMMA GEMM, one wave per tile -----------------
// 16 v_wmma per K=32 step (4 A-frags x 4 B-frags, fragment reuse).
// C[row,col] = act( sum_k A[row,k] * W~[k,col] + bias_node[row&1023] )
// W~[k,col] = wtrans ? W[col*ldw + k] : W[k*ldw + col]
// store:  C[(row*ldc + col) * ostride]   (ostride=3 for stacked logits)
__global__ __launch_bounds__(32)
void gemm64(const float* __restrict__ A, int lda, long bsA,
            const float* __restrict__ W, int ldw, long bsW, int wtrans,
            float* __restrict__ Cp, int ldc, long bsC, int ostride,
            int K, const float* __restrict__ bias_node, int act)
{
  __shared__ _Float16 Asm[64*32];   // 64 rows x 32 k
  __shared__ _Float16 Bt[64*40];    // 64 n-rows x 32 k (padded stride 40)
  const int lane = threadIdx.x;
  A  += (long)blockIdx.z * bsA;
  W  += (long)blockIdx.z * bsW;
  Cp += (long)blockIdx.z * bsC;
  const int n0 = blockIdx.x * 64;
  const int m0 = blockIdx.y * 64;
  v8f acc[16] = {};
  for (int k0 = 0; k0 < K; k0 += 32) {
    if (k0 + 32 < K) __builtin_prefetch(&A[(long)m0*lda + k0 + 32 + lane], 0, 0);
#pragma unroll 8
    for (int t = 0; t < 64; ++t)
      Asm[t*32 + lane] = (_Float16)A[(long)(m0 + t)*lda + k0 + lane];
    if (wtrans) {
#pragma unroll 8
      for (int t = 0; t < 64; ++t)
        Bt[t*40 + lane] = (_Float16)W[(long)(n0 + t)*ldw + k0 + lane];
    } else {
#pragma unroll 8
      for (int t = 0; t < 64; ++t) {
        int k = t >> 1;
        int n = ((t & 1) << 5) + lane;
        Bt[n*40 + k] = (_Float16)W[(long)(k0 + k)*ldw + n0 + n];
      }
    }
    __syncthreads();
    v16h af[4];
#pragma unroll
    for (int ma = 0; ma < 4; ++ma) af[ma] = frag_a_lds(Asm + ma*16*32, lane, 32);
#pragma unroll
    for (int nb = 0; nb < 4; ++nb) {
      v16h bf = frag_b_lds(Bt + nb*16*40, lane, 40);
#pragma unroll
      for (int ma = 0; ma < 4; ++ma)
        acc[ma*4 + nb] = __builtin_amdgcn_wmma_f32_16x16x32_f16(
            false, af[ma], false, bf, (short)0, acc[ma*4 + nb], false, false);
    }
    __syncthreads();
  }
  const int n  = lane & 15;
  const int mb = (lane < 16) ? 0 : 8;
#pragma unroll
  for (int ma = 0; ma < 4; ++ma)
#pragma unroll
    for (int nb = 0; nb < 4; ++nb)
#pragma unroll
      for (int v = 0; v < 8; ++v) {
        int row = m0 + ma*16 + mb + v;
        int col = n0 + nb*16 + n;
        float val = acc[ma*4 + nb][v];
        if (bias_node) val += bias_node[row & (Nn - 1)];
        if (act == 1)      val = 1.0f / (1.0f + __expf(-val));
        else if (act == 2) val = tanhf(val);
        Cp[((long)row*ldc + col) * (long)ostride] = val;
      }
}

// ---------------- per-node attention scores  s_src / s_dst ---------------------
__global__ void gat_scores(const float* __restrict__ hh,
                           const float* __restrict__ att_src,
                           const float* __restrict__ att_dst,
                           float* __restrict__ ssrc, float* __restrict__ sdst)
{
  int idx = blockIdx.x * blockDim.x + threadIdx.x;   // (b*N+n)*H + h
  if (idx >= BN*Hh) return;
  int h  = idx & (Hh - 1);
  long bn = idx >> 2;
  const float* row = hh + bn*HIN + h*Cc;
  float s0 = 0.f, s1 = 0.f;
#pragma unroll
  for (int c = 0; c < Cc; ++c) {
    float v = row[c];
    s0 += v * att_src[h*Cc + c];
    s1 += v * att_dst[h*Cc + c];
  }
  ssrc[idx] = s0;
  sdst[idx] = s1;
}

// ---------------- softmax row stats: max_j, 1/sum_j exp ------------------------
__global__ __launch_bounds__(32)
void attn_stats(const float* __restrict__ a,
                const float* __restrict__ ssrc,
                const float* __restrict__ sdst,
                float* __restrict__ rowmax, float* __restrict__ invden)
{
  const int lane = threadIdx.x;
  const int bi = blockIdx.x;             // b*N + i
  const int b  = bi >> 10;
  const float* arow = a + (long)bi*Nn;
  float sr[Hh], m[Hh], s[Hh];
#pragma unroll
  for (int h = 0; h < Hh; ++h) { sr[h] = ssrc[bi*Hh + h]; m[h] = -3.0e38f; s[h] = 0.f; }
  for (int j = lane; j < Nn; j += 32) {
    float aij  = arow[j];
    float mask = NEGINF * (1.0f - aij);
    const float* sd = sdst + ((long)(b*Nn + j))*Hh;
#pragma unroll
    for (int h = 0; h < Hh; ++h) {
      float e = sr[h] + sd[h];
      e = (e > 0.f) ? e : 0.2f*e;        // leaky_relu(.,0.2)
      e += mask;
      float mn = fmaxf(m[h], e);
      s[h] = s[h]*__expf(m[h] - mn) + __expf(e - mn);
      m[h] = mn;
    }
  }
#pragma unroll
  for (int h = 0; h < Hh; ++h) {
    float mm = m[h], ss = s[h];
    for (int off = 16; off >= 1; off >>= 1) {
      float mo = __shfl_xor(mm, off, 32);
      float so = __shfl_xor(ss, off, 32);
      float mn = fmaxf(mm, mo);
      ss = ss*__expf(mm - mn) + so*__expf(mo - mn);
      mm = mn;
    }
    if (lane == 0) { rowmax[bi*Hh + h] = mm; invden[bi*Hh + h] = 1.0f/ss; }
  }
}

// -------- fused softmax-numerator + WMMA aggregation: conv = attn @ hh ---------
// one wave per (b, h, 16-row i-tile); K-loop over j in chunks of 32.
// adjacency tile is DMA'd into LDS with GLOBAL_LOAD_ASYNC_TO_LDS_B32 (ASYNCcnt).
__global__ __launch_bounds__(32)
void attn_aggregate(const float* __restrict__ a,
                    const float* __restrict__ hh,
                    const float* __restrict__ ssrc,
                    const float* __restrict__ sdst,
                    const float* __restrict__ rowmax,
                    const float* __restrict__ invden,
                    const float* __restrict__ bias_gat,
                    float* __restrict__ conv)
{
  __shared__ float    Atile[16*32];      // async-DMA'd adjacency tile
  __shared__ _Float16 Psm[16*32];        // exp tile, rows i, cols j
  __shared__ _Float16 Bt[32*40];         // hh chunk transposed: [c][j]
  __shared__ float srcs[16], maxs[16], invs[16];
  const int lane = threadIdx.x;
  const int i0 = blockIdx.x * 16;
  const int h  = blockIdx.y;
  const int b  = blockIdx.z;
  if (lane < 16) {
    int bi = b*Nn + i0 + lane;
    srcs[lane] = ssrc[bi*Hh + h];
    maxs[lane] = rowmax[bi*Hh + h];
    invs[lane] = invden[bi*Hh + h];
  }
  __syncthreads();
  v8f acc0 = {}, acc1 = {};
  for (int j0 = 0; j0 < Nn; j0 += 32) {
    // --- async DMA the 16x32 f32 adjacency tile into LDS ---
    {
      const float* gp0 = a + ((long)(b*Nn + i0))*Nn + j0 + lane;
#pragma unroll
      for (int t = 0; t < 16; ++t) {
        unsigned lofs = (unsigned)(size_t)(&Atile[t*32 + lane]);
        const float* gp = gp0 + (long)t*Nn;
        asm volatile("global_load_async_to_lds_b32 %0, %1, off"
                     :: "v"(lofs), "v"(gp) : "memory");
      }
      asm volatile("s_wait_asynccnt 0x0" ::: "memory");
    }
    float sd = sdst[((long)(b*Nn + j0 + lane))*Hh + h];
#pragma unroll 2
    for (int t = 0; t < 16; ++t) {
      float aij = Atile[t*32 + lane];
      float e = srcs[t] + sd;
      e = (e > 0.f) ? e : 0.2f*e;
      e += NEGINF * (1.0f - aij);
      Psm[t*32 + lane] = (_Float16)__expf(e - maxs[t]);
    }
#pragma unroll 4
    for (int t = 0; t < 32; ++t)         // Bt[c=lane][j=t]
      Bt[lane*40 + t] = (_Float16)hh[((long)(b*Nn + j0 + t))*HIN + h*Cc + lane];
    __syncthreads();
    v16h af = frag_a_lds(Psm, lane, 32);
    v16h b0 = frag_b_lds(Bt,         lane, 40);
    v16h b1 = frag_b_lds(Bt + 16*40, lane, 40);
    acc0 = __builtin_amdgcn_wmma_f32_16x16x32_f16(false, af, false, b0, (short)0, acc0, false, false);
    acc1 = __builtin_amdgcn_wmma_f32_16x16x32_f16(false, af, false, b1, (short)0, acc1, false, false);
    __syncthreads();
  }
  const int n  = lane & 15;
  const int mb = (lane < 16) ? 0 : 8;
#pragma unroll
  for (int v = 0; v < 8; ++v) {
    int i = i0 + mb + v;
    float inv = invs[mb + v];
    int col = h*Cc + n;
    long base = ((long)(b*Nn + i))*HIN;
    conv[base + col]      = acc0[v]*inv + bias_gat[col];
    conv[base + col + 16] = acc1[v]*inv + bias_gat[col + 16];
  }
}

// ---------------- elementwise glue ---------------------------------------------
__global__ void build_cu(const float* __restrict__ conv, const float* __restrict__ h,
                         float* __restrict__ cu)
{
  long idx = (long)blockIdx.x*256 + threadIdx.x;
  if (idx >= (long)BN*256) return;
  long r = idx >> 8; int c = (int)(idx & 255);
  cu[idx] = (c < HIN) ? conv[r*HIN + c] : h[r*HID + (c - HIN)];
}
__global__ void build_cu2(const float* __restrict__ conv, const float* __restrict__ h,
                          const float* __restrict__ rg, float* __restrict__ cu2)
{
  long idx = (long)blockIdx.x*256 + threadIdx.x;
  if (idx >= (long)BN*256) return;
  long r = idx >> 8; int c = (int)(idx & 255);
  cu2[idx] = (c < HIN) ? conv[r*HIN + c]
                       : rg[r*HID + (c - HIN)] * h[r*HID + (c - HIN)];
}
__global__ void finish_h(const float* __restrict__ u, const float* __restrict__ h,
                         const float* __restrict__ cg, float* __restrict__ hp,
                         float* __restrict__ out_hp)
{
  long idx = (long)blockIdx.x*256 + threadIdx.x;
  if (idx >= (long)BN*HID) return;
  float uu = u[idx];
  float v = uu*h[idx] + (1.0f - uu)*cg[idx];
  hp[idx] = v; out_hp[idx] = v;
}

// ---------------- driver -------------------------------------------------------
extern "C" void kernel_launch(void* const* d_in, const int* in_sizes, int n_in,
                              void* d_out, int out_size, void* d_ws, size_t ws_size,
                              hipStream_t stream)
{
  const float* x        = (const float*)d_in[0];
  const float* a        = (const float*)d_in[1];
  const float* h        = (const float*)d_in[2];
  const float* Wk       = (const float*)d_in[3];
  const float* att_src  = (const float*)d_in[4];
  const float* att_dst  = (const float*)d_in[5];
  const float* bias_gat = (const float*)d_in[6];
  const float* b_u      = (const float*)d_in[7];
  const float* b_r      = (const float*)d_in[8];
  const float* b_c      = (const float*)d_in[9];
  const float* W_u      = (const float*)d_in[10];
  const float* W_r      = (const float*)d_in[11];
  const float* W_c      = (const float*)d_in[12];
  const float* R_p      = (const float*)d_in[13];
  const float* R_mu     = (const float*)d_in[14];
  const float* R_s      = (const float*)d_in[15];

  float* ws   = (float*)d_ws;
  float* hh   = ws; ws += (long)BN*HIN;
  float* ssrc = ws; ws += BN*Hh;
  float* sdst = ws; ws += BN*Hh;
  float* rmax = ws; ws += BN*Hh;
  float* rinv = ws; ws += BN*Hh;
  float* conv = ws; ws += (long)BN*HIN;
  float* cu   = ws; ws += (long)BN*256;
  float* ug   = ws; ws += (long)BN*HID;
  float* rg   = ws; ws += (long)BN*HID;
  float* cu2  = ws; ws += (long)BN*256;
  float* cg   = ws; ws += (long)BN*HID;
  float* hp   = ws; ws += (long)BN*HID;
  float* t0   = ws; ws += (long)BN*HID;
  float* t1   = ws; ws += (long)BN*HID;
  float* t2   = ws; ws += (long)BN*HID;

  float* out_logits = (float*)d_out;
  float* out_hp     = out_logits + (long)Bb*Nn*Nn*3;

  // 1. hh = x @ Wk    [8192,64]x[64,128]
  gemm64<<<dim3(HIN/64, BN/64, 1), 32, 0, stream>>>(
      x, Ff, 0, Wk, HIN, 0, 0, hh, HIN, 0, 1, Ff, nullptr, 0);
  // 2. per-node scores
  gat_scores<<<(BN*Hh + 255)/256, 256, 0, stream>>>(hh, att_src, att_dst, ssrc, sdst);
  // 3. softmax stats per (b,i,h)
  attn_stats<<<BN, 32, 0, stream>>>(a, ssrc, sdst, rmax, rinv);
  // 4. conv = softmax(e) @ hh + bias  (fused WMMA + async-LDS DMA)
  attn_aggregate<<<dim3(Nn/16, Hh, Bb), 32, 0, stream>>>(
      a, hh, ssrc, sdst, rmax, rinv, bias_gat, conv);
  // 5. GRU gates
  build_cu<<<(int)(((long)BN*256 + 255)/256), 256, 0, stream>>>(conv, h, cu);
  gemm64<<<dim3(HID/64, BN/64, 1), 32, 0, stream>>>(
      cu, 256, 0, W_u, HID, 0, 0, ug, HID, 0, 1, 256, b_u, 1);
  gemm64<<<dim3(HID/64, BN/64, 1), 32, 0, stream>>>(
      cu, 256, 0, W_r, HID, 0, 0, rg, HID, 0, 1, 256, b_r, 1);
  build_cu2<<<(int)(((long)BN*256 + 255)/256), 256, 0, stream>>>(conv, h, rg, cu2);
  gemm64<<<dim3(HID/64, BN/64, 1), 32, 0, stream>>>(
      cu2, 256, 0, W_c, HID, 0, 0, cg, HID, 0, 1, 256, b_c, 2);
  finish_h<<<(int)(((long)BN*HID + 255)/256), 256, 0, stream>>>(ug, h, cg, hp, out_hp);
  // 6. t_k = h' @ R_k
  gemm64<<<dim3(HID/64, BN/64, 1), 32, 0, stream>>>(
      hp, HID, 0, R_p,  HID, 0, 0, t0, HID, 0, 1, HID, nullptr, 0);
  gemm64<<<dim3(HID/64, BN/64, 1), 32, 0, stream>>>(
      hp, HID, 0, R_mu, HID, 0, 0, t1, HID, 0, 1, HID, nullptr, 0);
  gemm64<<<dim3(HID/64, BN/64, 1), 32, 0, stream>>>(
      hp, HID, 0, R_s,  HID, 0, 0, t2, HID, 0, 1, HID, nullptr, 0);
  // 7. logits[...,k] = t_k @ h'^T  (batched over b, strided store into [B,N,N,3])
  gemm64<<<dim3(Nn/64, Nn/64, Bb), 32, 0, stream>>>(
      t0, HID, (long)Nn*HID, hp, HID, (long)Nn*HID, 1,
      out_logits + 0, Nn, (long)Nn*Nn*3, 3, HID, nullptr, 0);
  gemm64<<<dim3(Nn/64, Nn/64, Bb), 32, 0, stream>>>(
      t1, HID, (long)Nn*HID, hp, HID, (long)Nn*HID, 1,
      out_logits + 1, Nn, (long)Nn*Nn*3, 3, HID, nullptr, 0);
  gemm64<<<dim3(Nn/64, Nn/64, Bb), 32, 0, stream>>>(
      t2, HID, (long)Nn*HID, hp, HID, (long)Nn*HID, 1,
      out_logits + 2, Nn, (long)Nn*Nn*3, 3, HID, nullptr, 0);
}
